// GaussianReadout_12670153523326
// MI455X (gfx1250) — matrix operand order repaired and checked by reading, not compiled, not run on hardware
//
#include <hip/hip_runtime.h>

typedef __bf16 bf16;
typedef __attribute__((ext_vector_type(16))) __bf16 v16bf;
typedef __attribute__((ext_vector_type(8)))  __bf16 v8bf;
typedef __attribute__((ext_vector_type(8)))  float  v8f;

namespace {
constexpr int kB   = 32;
constexpr int kC   = 256;
constexpr int kH   = 36;
constexpr int kW   = 64;
constexpr int kHW  = kH * kW;           // 2304
constexpr int kN   = 4096;

constexpr int KSTEP   = 32;             // K per stage (one WMMA-K)
constexpr int KPAD    = 40;             // LDS row stride in bf16 (80B: 16B aligned, bank-conflict-free)
constexpr int MTILE   = 128;            // n rows per workgroup
constexpr int NTILE   = 256;            // c cols per workgroup (all of C)
constexpr int NSTAGES = kHW / KSTEP;    // 72
}

// ---------------------------------------------------------------------------
// Kernel 1: feat fp32 [B,C,H,W] -> bf16 in the SAME layout ([B,C,HW] row-major,
// hw contiguous). This is exactly the [N][K] orientation the WMMA B-fragment
// wants, so no transpose is needed.
// ---------------------------------------------------------------------------
__global__ __launch_bounds__(256) void gr_feat_to_bf16(const float* __restrict__ feat,
                                                       bf16* __restrict__ out) {
    const size_t i = ((size_t)blockIdx.x * 256 + threadIdx.x) * 8;
    const float4 f0 = *(const float4*)(feat + i);
    const float4 f1 = *(const float4*)(feat + i + 4);
    v8bf o;
    o[0] = (bf16)f0.x; o[1] = (bf16)f0.y; o[2] = (bf16)f0.z; o[3] = (bf16)f0.w;
    o[4] = (bf16)f1.x; o[5] = (bf16)f1.y; o[6] = (bf16)f1.z; o[7] = (bf16)f1.w;
    *(v8bf*)(out + i) = o;
}

// ---------------------------------------------------------------------------
// Kernel 2: Gaussian masks G[n, hw] in bf16, row-major (hw contiguous) --
// the [M][K] orientation the WMMA A-fragment wants.
// ---------------------------------------------------------------------------
__global__ __launch_bounds__(256) void gr_gauss_masks(const float* __restrict__ mu,
                                                      const float* __restrict__ logsx,
                                                      const float* __restrict__ logsy,
                                                      const float* __restrict__ rho,
                                                      bf16* __restrict__ G) {
    const int n = blockIdx.x;
    const float mux = mu[2 * n], muy = mu[2 * n + 1];
    const float isx = 1.0f / (expf(logsx[n]) + 1e-6f);
    const float isy = 1.0f / (expf(logsy[n]) + 1e-6f);
    const float r   = tanhf(rho[n]);
    const float nid = -1.0f / (2.0f * (1.0f - r * r + 1e-6f));
    bf16* row = G + (size_t)n * kHW;
    for (int i = threadIdx.x; i < kHW; i += 256) {
        const int h = i >> 6, w = i & 63;
        const float X = (float)w * (2.0f / 63.0f) - 1.0f;
        const float Y = (float)h * (2.0f / 35.0f) - 1.0f;
        const float xc = (X - mux) * isx;
        const float yc = (Y - muy) * isy;
        const float A  = xc * xc + yc * yc - 2.0f * r * xc * yc;
        row[i] = (bf16)expf(A * nid);
    }
}

// ---------------------------------------------------------------------------
// Kernel 3: fused GEMM + weight-dot.
//   pooled_tile = G[nblk, :] @ feat[b]^T   (K = HW = 2304, bf16 WMMA, f32 acc)
//   out[b, n]   = sum_c pooled[n, c] * weight[n, c]   (fused epilogue)
// Grid: (kN/MTILE, kB). Block: 256 threads = 8 waves, arranged 2(n) x 4(c),
// each wave computing a 64x64 sub-tile = 4x4 WMMA accumulator tiles.
// ---------------------------------------------------------------------------
__global__ __launch_bounds__(256) void gr_gemm_fused(const bf16* __restrict__ G,
                                                     const bf16* __restrict__ F,
                                                     const float* __restrict__ Wt,
                                                     float* __restrict__ out) {
    __shared__ __align__(16) bf16 As[2][MTILE][KPAD];
    __shared__ __align__(16) bf16 Bs[2][NTILE][KPAD];
    __shared__ float outred[MTILE];

    const int tid  = threadIdx.x;
    const int lane = tid & 31;
    const int wave = tid >> 5;
    const int wn   = wave >> 2;   // 0..1 : n-subtile
    const int wc   = wave & 3;    // 0..3 : c-subtile
    const int nblk = blockIdx.x;  // 0..31
    const int b    = blockIdx.y;  // 0..31
    const int nbase = nblk * MTILE;

    const bf16* __restrict__ Abase = G + (size_t)nbase * kHW;
    const bf16* __restrict__ Bbase = F + (size_t)b * kC * kHW;

    // ---- fragment addressing (CDNA5 wave32 WMMA layouts) ----
    const int half  = lane >> 4;          // 0: lanes 0-15, 1: lanes 16-31
    const int lidx  = lane & 15;
    const int arow  = wn * 64 + lidx;     // A: M row per lane
    const int akoff = half * 8;           // A: K runs {akoff..+7, akoff+16..+23}
    const int brow  = wc * 64 + lidx;     // B: N col per lane (c index)
    const int bkoff = half * 8;           // B: K runs {bkoff..+7, bkoff+16..+23}

    v8f acc[4][4];
#pragma unroll
    for (int mt = 0; mt < 4; ++mt)
#pragma unroll
        for (int ct = 0; ct < 4; ++ct)
            acc[mt][ct] = (v8f){0.f, 0.f, 0.f, 0.f, 0.f, 0.f, 0.f, 0.f};

    // ---- prologue: stage 0 into LDS buffer 0 ----
#pragma unroll
    for (int i = 0; i < 2; ++i) {                 // A: 128x32 = 512 b128 chunks
        const int c  = tid + 256 * i;
        const int rw = c >> 2, kc = (c & 3) * 8;
        *(uint4*)&As[0][rw][kc] = *(const uint4*)(Abase + (size_t)rw * kHW + kc);
    }
#pragma unroll
    for (int i = 0; i < 4; ++i) {                 // B: 256x32 = 1024 b128 chunks
        const int c  = tid + 256 * i;
        const int rw = c >> 2, kc = (c & 3) * 8;
        *(uint4*)&Bs[0][rw][kc] = *(const uint4*)(Bbase + (size_t)rw * kHW + kc);
    }
    __syncthreads();

    // ---- main K loop, double buffered, one barrier per stage ----
    for (int ks = 0; ks < NSTAGES; ++ks) {
        const int buf   = ks & 1;
        const int knext = (ks + 1) * KSTEP;
        uint4 sa[2], sb[4];
        if (ks + 1 < NSTAGES) {
#pragma unroll
            for (int i = 0; i < 2; ++i) {
                const int c  = tid + 256 * i;
                const int rw = c >> 2, kc = (c & 3) * 8;
                sa[i] = *(const uint4*)(Abase + (size_t)rw * kHW + knext + kc);
            }
#pragma unroll
            for (int i = 0; i < 4; ++i) {
                const int c  = tid + 256 * i;
                const int rw = c >> 2, kc = (c & 3) * 8;
                sb[i] = *(const uint4*)(Bbase + (size_t)rw * kHW + knext + kc);
            }
        }

        // fragments + 16 WMMAs on current buffer
        v16bf afrag[4];
#pragma unroll
        for (int mt = 0; mt < 4; ++mt) {
            const bf16* p = &As[buf][arow + mt * 16][akoff];
            const v8bf lo = *(const v8bf*)p;
            const v8bf hi = *(const v8bf*)(p + 16);
            afrag[mt] = __builtin_shufflevector(lo, hi, 0, 1, 2, 3, 4, 5, 6, 7,
                                                8, 9, 10, 11, 12, 13, 14, 15);
        }
#pragma unroll
        for (int ct = 0; ct < 4; ++ct) {
            const bf16* p = &Bs[buf][brow + ct * 16][bkoff];
            const v8bf lo = *(const v8bf*)p;
            const v8bf hi = *(const v8bf*)(p + 16);
            const v16bf bfrag = __builtin_shufflevector(lo, hi, 0, 1, 2, 3, 4, 5, 6, 7,
                                                        8, 9, 10, 11, 12, 13, 14, 15);
#pragma unroll
            for (int mt = 0; mt < 4; ++mt)
                acc[mt][ct] = __builtin_amdgcn_wmma_f32_16x16x32_bf16(
                    false, afrag[mt], false, bfrag, (short)0, acc[mt][ct], false, false);
        }

        if (ks + 1 < NSTAGES) {
            const int nb = buf ^ 1;
#pragma unroll
            for (int i = 0; i < 2; ++i) {
                const int c  = tid + 256 * i;
                const int rw = c >> 2, kc = (c & 3) * 8;
                *(uint4*)&As[nb][rw][kc] = sa[i];
            }
#pragma unroll
            for (int i = 0; i < 4; ++i) {
                const int c  = tid + 256 * i;
                const int rw = c >> 2, kc = (c & 3) * 8;
                *(uint4*)&Bs[nb][rw][kc] = sb[i];
            }
            __syncthreads();
        }
    }

    // ---- fused epilogue: out[b,n] = sum_c acc * weight[n,c] ----
    if (tid < MTILE) outred[tid] = 0.0f;
    __syncthreads();

#pragma unroll
    for (int mt = 0; mt < 4; ++mt) {
#pragma unroll
        for (int r = 0; r < 8; ++r) {
            // C/D layout: VGPR r, lanes 0-15 -> M=r, lanes 16-31 -> M=r+8; N = lane&15
            const int rowl = wn * 64 + mt * 16 + r + half * 8;
            const int rowg = nbase + rowl;
            float s = 0.f;
#pragma unroll
            for (int ct = 0; ct < 4; ++ct) {
                const int col = wc * 64 + ct * 16 + lidx;
                s += acc[mt][ct][r] * Wt[(size_t)rowg * kC + col];
            }
            // reduce across the 16 lanes holding this row's columns
            s += __shfl_xor(s, 1, 32);
            s += __shfl_xor(s, 2, 32);
            s += __shfl_xor(s, 4, 32);
            s += __shfl_xor(s, 8, 32);
            if (lidx == 0) atomicAdd(&outred[rowl], s);   // combine the 4 c-waves
        }
    }
    __syncthreads();
    if (tid < MTILE) out[(size_t)b * kN + nbase + tid] = outred[tid];
}

// ---------------------------------------------------------------------------
// Host-side launcher
// ---------------------------------------------------------------------------
extern "C" void kernel_launch(void* const* d_in, const int* in_sizes, int n_in,
                              void* d_out, int out_size, void* d_ws, size_t ws_size,
                              hipStream_t stream) {
    const float* feat   = (const float*)d_in[0];   // [32,256,36,64]
    const float* mu     = (const float*)d_in[1];   // [4096,2]
    const float* logsx  = (const float*)d_in[2];   // [4096]
    const float* logsy  = (const float*)d_in[3];   // [4096]
    const float* rho    = (const float*)d_in[4];   // [4096]
    const float* weight = (const float*)d_in[5];   // [4096,256]
    float* out = (float*)d_out;                    // [32,4096]

    // workspace: feat bf16 [B,C,HW] then G bf16 [n,HW]  (~54 MB total)
    bf16* featb = (bf16*)d_ws;
    bf16* G     = (bf16*)((char*)d_ws + (size_t)kB * kC * kHW * sizeof(bf16));

    const int convBlocks = (kB * kC * kHW) / (256 * 8);      // 9216
    gr_feat_to_bf16<<<dim3(convBlocks), dim3(256), 0, stream>>>(feat, featb);
    gr_gauss_masks<<<dim3(kN), dim3(256), 0, stream>>>(mu, logsx, logsy, rho, G);
    gr_gemm_fused<<<dim3(kN / MTILE, kB), dim3(256), 0, stream>>>(G, featb, weight, out);
}
